// BaseModel_11020886081610
// MI455X (gfx1250) — compile-verified, matrix operand order checked by reading, not run on hardware
//
#include <hip/hip_runtime.h>
#include <math.h>
#include <stdint.h>

// ---------------------------------------------------------------------------
// Model constants
// ---------------------------------------------------------------------------
#define BB     256
#define SEEDN  120
#define PREDN  24
#define DD     126
#define HH     1024
#define JJ     14
#define JDD    9
#define HEADSN 4
#define DHN    256      // HH / HEADSN
#define SPLHN  128
#define FFN    61       // rfft bins of 120
#define KPAD   128      // D padded
#define SPLK   1056     // (H + JD)=1033 padded to mult of 32
#define LPAD   40       // LDS stage row pitch in halves (80B)
#define HHP    136      // LDS hh-tile row pitch in halves (272B)

typedef _Float16 f16;
typedef _Float16 h8   __attribute__((ext_vector_type(8)));
typedef _Float16 v16h __attribute__((ext_vector_type(16)));
typedef float    v8f  __attribute__((ext_vector_type(8)));

// ---------------------------------------------------------------------------
// WMMA fragment load, row-major source: lane l holds row (l&15),
// elems 0..7 = K[k0 + (l>>4)*8 ..], elems 8..15 = same + 16  (two 16B loads)
// -> CDNA5 16-bit A/B 16x32 VGPR layout.
// ---------------------------------------------------------------------------
__device__ __forceinline__ v16h frag_load(const f16* __restrict__ base, int ld,
                                          int row, int k0, int hsel) {
    const f16* p = base + (size_t)row * ld + k0 + hsel * 8;
    h8 lo = *(const h8*)p;
    h8 hi = *(const h8*)(p + 16);
    return __builtin_shufflevector(lo, hi, 0,1,2,3,4,5,6,7,8,9,10,11,12,13,14,15);
}

__device__ __forceinline__ v16h lds_frag(const f16* base, int row, int k0,
                                         int hsel, int pitch) {
    const f16* p = base + row * pitch + k0 + hsel * 8;
    h8 lo = *(const h8*)p;            // ds_load_b128
    h8 hi = *(const h8*)(p + 16);     // ds_load_b128
    return __builtin_shufflevector(lo, hi, 0,1,2,3,4,5,6,7,8,9,10,11,12,13,14,15);
}

// CDNA5 async copy global -> LDS (16B per lane), tracked by ASYNCcnt.
__device__ __forceinline__ void async_load_b128(uint32_t lds_off, const f16* gptr) {
    asm volatile("global_load_async_to_lds_b128 %0, %1, off"
                 :: "v"(lds_off), "v"(gptr) : "memory");
}

__device__ __forceinline__ v8f wmma_step(v16h a, v16h b, v8f c) {
    return __builtin_amdgcn_wmma_f32_16x16x32_f16(false, a, false, b, (short)0, c,
                                                  false, false);
}

// ---------------------------------------------------------------------------
// Shared staged-GEMM inner loop: double-buffered async global->LDS copies,
// block tile 64(M) x 128(N), 8 waves, wave tile 32x32 = 2x2 wmma 16x16x32.
// Requirements: M % 64 == 0, N % 128 == 0, K % 32 == 0.
// ---------------------------------------------------------------------------
__device__ __forceinline__ void staged_mm(
    const f16* __restrict__ A, int lda, const f16* __restrict__ W, int ldw, int K,
    int row0g, int col0g, int tid,
    f16 (&smA)[2][64 * LPAD], f16 (&smW)[2][128 * LPAD], v8f (&acc)[2][2])
{
    const int wave = tid >> 5, lane = tid & 31;
    const int wr = wave & 1, wc = wave >> 1;
    const int lrow = lane & 15, hsel = lane >> 4;
    const int ra = tid >> 2, quad = tid & 3;     // 4 lanes per 64B row
    const f16* gA  = A + (size_t)(row0g + ra) * lda + quad * 8;
    const f16* gW0 = W + (size_t)(col0g + ra) * ldw + quad * 8;
    const f16* gW1 = W + (size_t)(col0g + 64 + ra) * ldw + quad * 8;

    auto issue = [&](int k0, int buf) {
        async_load_b128((uint32_t)(uintptr_t)&smA[buf][ra * LPAD + quad * 8], gA + k0);
        async_load_b128((uint32_t)(uintptr_t)&smW[buf][ra * LPAD + quad * 8], gW0 + k0);
        async_load_b128((uint32_t)(uintptr_t)&smW[buf][(64 + ra) * LPAD + quad * 8], gW1 + k0);
    };

    const int KT = K >> 5;
    issue(0, 0);
    for (int kt = 0; kt < KT; ++kt) {
        const int cur = kt & 1;
        if (kt + 1 < KT) {
            issue((kt + 1) << 5, cur ^ 1);
            asm volatile("s_wait_asynccnt 0x3" ::: "memory");   // tile kt landed
        } else {
            asm volatile("s_wait_asynccnt 0x0" ::: "memory");
        }
        __syncthreads();
        const f16* bA = &smA[cur][0];
        const f16* bW = &smW[cur][0];
        v16h a0 = lds_frag(bA, wr * 32 + lrow,      0, hsel, LPAD);
        v16h a1 = lds_frag(bA, wr * 32 + 16 + lrow, 0, hsel, LPAD);
        v16h b0 = lds_frag(bW, wc * 32 + lrow,      0, hsel, LPAD);
        v16h b1 = lds_frag(bW, wc * 32 + 16 + lrow, 0, hsel, LPAD);
        acc[0][0] = wmma_step(a0, b0, acc[0][0]);
        acc[0][1] = wmma_step(a0, b1, acc[0][1]);
        acc[1][0] = wmma_step(a1, b0, acc[1][0]);
        acc[1][1] = wmma_step(a1, b1, acc[1][1]);
        __syncthreads();                                        // protect buffer reuse
    }
}

// Epilogue: bias -> act -> addmat; optional f32 / f16 stores.
// C/D layout: lane<16 -> N=lane, M=vgpr(0..7); lane>=16 -> N=lane-16, M=8+vgpr
__device__ __forceinline__ void gemm_epilogue(
    const v8f accs[2][2], int row0, int col0,
    const float* __restrict__ bias, const float* __restrict__ addmat, int ldadd,
    int act, float* __restrict__ out32, int ldc, f16* __restrict__ out16, int ldc16,
    int lane)
{
    const int nbase = lane & 15;
    const int moff  = (lane >> 4) * 8;
    #pragma unroll
    for (int r = 0; r < 2; ++r) {
        #pragma unroll
        for (int c = 0; c < 2; ++c) {
            const int n = col0 + 16 * c + nbase;
            const float bv = bias ? bias[n] : 0.0f;
            const int mrow0 = row0 + 16 * r + moff;
            #pragma unroll
            for (int v = 0; v < 8; ++v) {
                const int m = mrow0 + v;
                float val = accs[r][c][v] + bv;
                if (act == 1) val = fmaxf(val, 0.0f);
                if (addmat) val += addmat[(size_t)m * ldadd + n];
                if (out32)  out32[(size_t)m * ldc + n] = val;
                if (out16)  out16[(size_t)m * ldc16 + n] = (f16)val;
            }
        }
    }
}

// ---------------------------------------------------------------------------
// Generic staged GEMM kernel: out = act(A * W^T + bias) + addmat
// ---------------------------------------------------------------------------
__global__ __launch_bounds__(256) void gemm_wmma_lds(
    const f16* __restrict__ A, int lda,
    const f16* __restrict__ W, int ldw,
    int K,
    const float* __restrict__ bias,
    const float* __restrict__ addmat, int ldadd,
    int act,
    float* __restrict__ out32, int ldc,
    f16* __restrict__ out16, int ldc16)
{
    __shared__ __attribute__((aligned(16))) f16 smA[2][64 * LPAD];
    __shared__ __attribute__((aligned(16))) f16 smW[2][128 * LPAD];
    const int tid = threadIdx.x;
    const int row0g = blockIdx.y * 64;
    const int col0g = blockIdx.x * 128;
    v8f acc[2][2] = {};
    staged_mm(A, lda, W, ldw, K, row0g, col0g, tid, smA, smW, acc);
    const int wave = tid >> 5;
    gemm_epilogue(acc, row0g + (wave & 1) * 32, col0g + (wave >> 1) * 32,
                  bias, addmat, ldadd, act, out32, ldc, out16, ldc16, tid & 31);
}

// ---------------------------------------------------------------------------
// Fused GRU pair: blockIdx.z selects (A,W,bias,out,K) for gi / gh.
// N = 3*HH, M = BB.  Grid: (3*HH/128, BB/64, 2).
// ---------------------------------------------------------------------------
__global__ __launch_bounds__(256) void gemm_gru2_k(
    const f16* __restrict__ A0, const f16* __restrict__ W0,
    const float* __restrict__ b0, float* __restrict__ o0, int K0,
    const f16* __restrict__ A1, const f16* __restrict__ W1,
    const float* __restrict__ b1, float* __restrict__ o1, int K1)
{
    __shared__ __attribute__((aligned(16))) f16 smA[2][64 * LPAD];
    __shared__ __attribute__((aligned(16))) f16 smW[2][128 * LPAD];
    const f16* A; const f16* W; const float* bias; float* out; int K;
    if (blockIdx.z == 0) { A = A0; W = W0; bias = b0; out = o0; K = K0; }
    else                 { A = A1; W = W1; bias = b1; out = o1; K = K1; }
    const int tid = threadIdx.x;
    const int row0g = blockIdx.y * 64;
    const int col0g = blockIdx.x * 128;
    v8f acc[2][2] = {};
    staged_mm(A, K, W, K, K, row0g, col0g, tid, smA, smW, acc);
    const int wave = tid >> 5;
    gemm_epilogue(acc, row0g + (wave & 1) * 32, col0g + (wave >> 1) * 32,
                  bias, nullptr, 0, 0, out, 3 * HH, nullptr, 0, tid & 31);
}

// ---------------------------------------------------------------------------
// Fused SPL level kernel: one block = one joint x 64-row tile.
//   hh  = relu(hid @ W1[:, :1024]^T + par @ W1[:, 1024:1033]^T + b1)  (WMMA, staged)
//   out = tanh(hh @ W2^T + b2) -> xdelta[:, j*9 .. +9), preds16[j]     (WMMA from LDS)
// Grid: (BB/64, #joints at this depth).
// ---------------------------------------------------------------------------
struct JointSet { int j[8]; int p[8]; };

__global__ __launch_bounds__(256) void spl_depth_k(
    const f16* __restrict__ hid16,          // BB x HH
    const f16* __restrict__ splW1,          // JJ x 128 x SPLK
    const float* __restrict__ spl_b1,       // JJ x 128
    const f16* __restrict__ splW2,          // JJ x 16 x 128
    const float* __restrict__ spl_b2,       // JJ x 9
    f16* __restrict__ preds16,              // JJ x BB x 16
    float* __restrict__ xdelta,             // BB x DD
    JointSet js)
{
    __shared__ __attribute__((aligned(16))) f16 smA[2][64 * LPAD];
    __shared__ __attribute__((aligned(16))) f16 smW[2][128 * LPAD];
    __shared__ __attribute__((aligned(16))) f16 hhL[64 * HHP];
    __shared__ f16 parL[64 * 16];

    const int jj  = js.j[blockIdx.y];
    const int par = js.p[blockIdx.y];
    const int tid = threadIdx.x;
    const int wave = tid >> 5, lane = tid & 31;
    const int wr = wave & 1, wc = wave >> 1;
    const int row0g = blockIdx.x * 64;
    const int lrow = lane & 15, hsel = lane >> 4;

    // stage parent predictions (zeros for roots); visible after first barrier
    for (int i = tid; i < 64 * 16; i += 256) {
        int rr = i >> 4, cc = i & 15;
        f16 v = (f16)0.0f;
        if (par >= 0 && cc < JDD)
            v = preds16[((size_t)par * BB + row0g + rr) * 16 + cc];
        parL[i] = v;
    }

    // ---- layer 1: staged WMMA over K = 1024 (hid part of W1) ----
    const f16* W1 = splW1 + (size_t)jj * SPLHN * SPLK;
    v8f acc[2][2] = {};
    staged_mm(hid16, HH, W1, SPLK, HH, row0g, 0, tid, smA, smW, acc);

    // epilogue: bias + parent term (9 MACs) -> relu -> hh tile in LDS
    const int nbase = lane & 15;
    const int moff  = (lane >> 4) * 8;
    #pragma unroll
    for (int r = 0; r < 2; ++r) {
        #pragma unroll
        for (int c = 0; c < 2; ++c) {
            const int n = wc * 32 + 16 * c + nbase;          // 0..127
            const float bv = spl_b1[jj * SPLHN + n];
            const f16* w1b = W1 + (size_t)n * SPLK + HH;     // parent block of row n
            #pragma unroll
            for (int v = 0; v < 8; ++v) {
                const int mloc = wr * 32 + 16 * r + moff + v;
                float val = acc[r][c][v] + bv;
                #pragma unroll
                for (int q = 0; q < JDD; ++q)
                    val += (float)parL[mloc * 16 + q] * (float)w1b[q];
                val = fmaxf(val, 0.0f);
                hhL[mloc * HHP + n] = (f16)val;
            }
        }
    }
    __syncthreads();

    // ---- layer 2: out(64x16) = hh(64x128) @ W2(16x128)^T, waves 0..3 ----
    if (wave < 4) {
        const f16* W2 = splW2 + (size_t)jj * 16 * SPLHN;
        v8f a2 = {};
        #pragma unroll
        for (int k0 = 0; k0 < SPLHN; k0 += 32) {
            v16h af = lds_frag(&hhL[0], wave * 16 + lrow, k0, hsel, HHP);
            v16h bf = frag_load(W2, SPLHN, lrow, k0, hsel);
            a2 = wmma_step(af, bf, a2);
        }
        if (nbase < JDD) {                     // divergence only after WMMA
            const float bv = spl_b2[jj * JDD + nbase];
            #pragma unroll
            for (int v = 0; v < 8; ++v) {
                const int mloc = wave * 16 + moff + v;
                float val = tanhf(a2[v] + bv);
                xdelta[(size_t)(row0g + mloc) * DD + jj * JDD + nbase] = val;
                preds16[((size_t)jj * BB + row0g + mloc) * 16 + nbase] = (f16)val;
            }
        }
    }
}

// ---------------------------------------------------------------------------
// Elementwise / helper kernels
// ---------------------------------------------------------------------------
__global__ void convert_pad_k(const float* __restrict__ src, f16* __restrict__ dst,
                              int rows, int scols, int dld) {
    long i = (long)blockIdx.x * blockDim.x + threadIdx.x;
    long total = (long)rows * dld;
    if (i >= total) return;
    int r = (int)(i / dld), c = (int)(i % dld);
    dst[i] = (f16)((c < scols) ? src[(size_t)r * scols + c] : 0.0f);
}

// spl_W2 (J, JD, SPLH) -> (J, 16, SPLH) row-padded f16
__global__ void convert_w2_k(const float* __restrict__ src, f16* __restrict__ dst) {
    int i = blockIdx.x * blockDim.x + threadIdx.x;
    if (i >= JJ * 16 * SPLHN) return;
    int c = i & (SPLHN - 1);
    int r = i >> 7;
    int j = r >> 4, jr = r & 15;
    dst[i] = (f16)((jr < JDD) ? src[((size_t)j * JDD + jr) * SPLHN + c] : 0.0f);
}

__global__ void cos_table_k(float* __restrict__ tab) {
    int i = blockIdx.x * blockDim.x + threadIdx.x;
    if (i >= FFN * SEEDN) return;
    int f = i / SEEDN, t = i % SEEDN;
    tab[i] = cosf((2.0f * 3.14159265358979323846f * (float)f * (float)t) / (float)SEEDN);
}

// freq16[(b*61+f)*128 + d] = sum_t cos(2pi f t/120) * poses[b,t,d]
__global__ __launch_bounds__(128) void freq_k(const float* __restrict__ poses,
                                              const float* __restrict__ tab,
                                              f16* __restrict__ freq16) {
    int bf = blockIdx.x;
    int b = bf / FFN, f = bf % FFN;
    __shared__ float ct[SEEDN];
    if (threadIdx.x < SEEDN) ct[threadIdx.x] = tab[f * SEEDN + threadIdx.x];
    __syncthreads();
    int d = threadIdx.x;
    float acc = 0.0f;
    if (d < DD) {
        const float* p = poses + ((size_t)b * (SEEDN + PREDN)) * DD + d;
        #pragma unroll 4
        for (int t = 0; t < SEEDN; ++t) acc += ct[t] * p[(size_t)t * DD];
    }
    freq16[(size_t)bf * KPAD + d] = (f16)((d < DD) ? acc : 0.0f);
}

// X16[t][b][k] f16, k padded 126->128
__global__ void build_x16_k(const float* __restrict__ poses, f16* __restrict__ X16) {
    long i = (long)blockIdx.x * blockDim.x + threadIdx.x;
    if (i >= (long)SEEDN * BB * KPAD) return;
    int k = (int)(i & (KPAD - 1));
    long bt = i >> 7;
    int b = (int)(bt % BB);
    int t = (int)(bt / BB);
    float v = (k < DD) ? poses[((size_t)b * (SEEDN + PREDN) + t) * DD + k] : 0.0f;
    X16[i] = (f16)v;
}

__global__ void init_state_k(const float* __restrict__ poses,
                             float* __restrict__ h, f16* __restrict__ h16,
                             float* __restrict__ x, f16* __restrict__ x16) {
    int i = blockIdx.x * blockDim.x + threadIdx.x;
    if (i < BB * HH) { h[i] = 0.0f; h16[i] = (f16)0.0f; }
    if (i < BB * KPAD) {
        int b = i >> 7, k = i & (KPAD - 1);
        float v = (k < DD) ? poses[((size_t)b * (SEEDN + PREDN) + (SEEDN - 1)) * DD + k] : 0.0f;
        x16[i] = (f16)v;
        if (k < DD) x[(size_t)b * DD + k] = v;
    }
}

__global__ void gru_gates_k(const float* __restrict__ gi, const float* __restrict__ gh,
                            float* __restrict__ h, f16* __restrict__ h16) {
    int i = blockIdx.x * blockDim.x + threadIdx.x;
    if (i >= BB * HH) return;
    int b = i >> 10, c = i & (HH - 1);
    const float* gib = gi + (size_t)b * 3 * HH;
    const float* ghb = gh + (size_t)b * 3 * HH;
    float r = 1.0f / (1.0f + expf(-(gib[c] + ghb[c])));
    float z = 1.0f / (1.0f + expf(-(gib[c + HH] + ghb[c + HH])));
    float n = tanhf(gib[c + 2 * HH] + r * ghb[c + 2 * HH]);
    float hv = h[i];
    float hn = (1.0f - z) * n + z * hv;
    h[i] = hn;
    h16[i] = (f16)hn;
}

// Attention for one (batch-in-chunk, head):
//   mean_q(ctx)[d] = sum_k ( sum_q att[q,k] / 61 ) * v[k,d]
__global__ __launch_bounds__(256) void attention_k(const f16* __restrict__ qkv,
                                                   int chunk_b0,
                                                   float* __restrict__ ctx_mean) {
    int bl = blockIdx.x;
    int hd = blockIdx.y;
    int tid = threadIdx.x;
    __shared__ f16  kl[FFN * DHN];
    __shared__ float sc[FFN * FFN];
    __shared__ float w[FFN];
    const f16* base = qkv + (size_t)bl * FFN * 3 * HH;
    for (int i = tid; i < FFN * DHN; i += 256) {
        int p = i >> 8, d = i & (DHN - 1);
        kl[i] = base[(size_t)p * 3 * HH + HH + hd * DHN + d];
    }
    __syncthreads();
    const float scale = 0.0625f;   // 1/sqrt(256)
    for (int i = tid; i < FFN * FFN; i += 256) {
        int qi = i / FFN, kj = i % FFN;
        const f16* qrow = base + (size_t)qi * 3 * HH + hd * DHN;
        float acc = 0.0f;
        #pragma unroll 8
        for (int d = 0; d < DHN; ++d) acc += (float)qrow[d] * (float)kl[kj * DHN + d];
        sc[i] = acc * scale;
    }
    __syncthreads();
    if (tid < FFN) {
        float mx = -1e30f;
        for (int j = 0; j < FFN; ++j) mx = fmaxf(mx, sc[tid * FFN + j]);
        float s = 0.0f;
        for (int j = 0; j < FFN; ++j) { float e = expf(sc[tid * FFN + j] - mx); sc[tid * FFN + j] = e; s += e; }
        float inv = 1.0f / s;
        for (int j = 0; j < FFN; ++j) sc[tid * FFN + j] *= inv;
    }
    __syncthreads();
    if (tid < FFN) {
        float s = 0.0f;
        for (int q = 0; q < FFN; ++q) s += sc[q * FFN + tid];
        w[tid] = s / (float)FFN;
    }
    __syncthreads();
    if (tid < DHN) {
        float acc = 0.0f;
        for (int p = 0; p < FFN; ++p)
            acc += w[p] * (float)base[(size_t)p * 3 * HH + 2 * HH + hd * DHN + tid];
        ctx_mean[((size_t)(chunk_b0 + bl)) * HH + hd * DHN + tid] = acc;
    }
}

__global__ void x_update_k(float* __restrict__ x, f16* __restrict__ x16,
                           const float* __restrict__ xdelta,
                           float* __restrict__ out, int t) {
    int i = blockIdx.x * blockDim.x + threadIdx.x;
    if (i >= BB * DD) return;
    int b = i / DD, d = i % DD;
    float v = x[i] + xdelta[i];
    x[i] = v;
    x16[(size_t)b * KPAD + d] = (f16)v;
    out[((size_t)b * PREDN + t) * DD + d] = v;
}

// ---------------------------------------------------------------------------
// Host orchestration
// ---------------------------------------------------------------------------
static inline void launch_gemm(hipStream_t s, const f16* A, int lda, const f16* W, int ldw,
                               int M, int N, int K, const float* bias,
                               const float* addmat, int ldadd, int act,
                               float* o32, int ldc, f16* o16, int ldc16) {
    dim3 g(N / 128, M / 64);       // all call sites have N%128==0, M%64==0
    gemm_wmma_lds<<<g, 256, 0, s>>>(A, lda, W, ldw, K, bias, addmat, ldadd, act,
                                    o32, ldc, o16, ldc16);
}

extern "C" void kernel_launch(void* const* d_in, const int* in_sizes, int n_in,
                              void* d_out, int out_size, void* d_ws, size_t ws_size,
                              hipStream_t stream) {
    (void)in_sizes; (void)n_in; (void)out_size; (void)ws_size;
    const float* poses     = (const float*)d_in[0];
    const float* gru_Wih   = (const float*)d_in[1];
    const float* gru_Whh   = (const float*)d_in[2];
    const float* gru_bih   = (const float*)d_in[3];
    const float* gru_bhh   = (const float*)d_in[4];
    const float* pre_W     = (const float*)d_in[5];
    const float* pre_b     = (const float*)d_in[6];
    const float* fp_W      = (const float*)d_in[7];
    const float* fp_b      = (const float*)d_in[8];
    const float* inproj_W  = (const float*)d_in[9];
    const float* inproj_b  = (const float*)d_in[10];
    const float* outproj_W = (const float*)d_in[11];
    const float* outproj_b = (const float*)d_in[12];
    const float* spl_W1    = (const float*)d_in[13];
    const float* spl_b1    = (const float*)d_in[14];
    const float* spl_W2    = (const float*)d_in[15];
    const float* spl_b2    = (const float*)d_in[16];
    float* out = (float*)d_out;

    // ---- workspace carve (256B aligned) ----
    char* ws = (char*)d_ws;
    size_t off = 0;
    auto alloc = [&](size_t bytes) -> char* {
        char* p = ws + off;
        off = (off + bytes + 255) & ~(size_t)255;
        return p;
    };
    f16* Wih16    = (f16*)alloc((size_t)3*HH*KPAD * 2);
    f16* Whh16    = (f16*)alloc((size_t)3*HH*HH * 2);
    f16* pre16    = (f16*)alloc((size_t)HH*HH * 2);
    f16* fpW16    = (f16*)alloc((size_t)HH*KPAD * 2);
    f16* inproj16 = (f16*)alloc((size_t)3*HH*HH * 2);
    f16* outproj16= (f16*)alloc((size_t)HH*HH * 2);
    f16* splW1_16 = (f16*)alloc((size_t)JJ*SPLHN*SPLK * 2);
    f16* splW2_16 = (f16*)alloc((size_t)JJ*16*SPLHN * 2);
    float* costab = (float*)alloc((size_t)FFN*SEEDN * 4);
    f16* X16      = (f16*)alloc((size_t)SEEDN*BB*KPAD * 2);
    f16* freq16   = (f16*)alloc((size_t)BB*FFN*KPAD * 2);
    const int BC = 64;                       // attention batch chunk
    const int MC = BC * FFN;                 // 3904 rows (mult of 64)
    f16* feat16c  = (f16*)alloc((size_t)MC*HH * 2);
    f16* qkv16c   = (f16*)alloc((size_t)MC*3*HH * 2);
    float* ctxm   = (float*)alloc((size_t)BB*HH * 4);
    f16* ctxm16   = (f16*)alloc((size_t)BB*HH * 2);
    float* motion = (float*)alloc((size_t)BB*HH * 4);
    float* h      = (float*)alloc((size_t)BB*HH * 4);
    f16* h16      = (f16*)alloc((size_t)BB*HH * 2);
    float* gi     = (float*)alloc((size_t)BB*3*HH * 4);
    float* gh     = (float*)alloc((size_t)BB*3*HH * 4);
    f16* hid16    = (f16*)alloc((size_t)BB*HH * 2);
    f16* preds16  = (f16*)alloc((size_t)JJ*BB*16 * 2);
    float* xdelta = (float*)alloc((size_t)BB*DD * 4);
    float* x      = (float*)alloc((size_t)BB*DD * 4);
    f16* x16      = (f16*)alloc((size_t)BB*KPAD * 2);

    auto g1 = [](long n) { return dim3((unsigned)((n + 255) / 256)); };

    // ---- weight conversion to f16 (K padded to mult of 32) ----
    convert_pad_k<<<g1((long)3*HH*KPAD), 256, 0, stream>>>(gru_Wih, Wih16, 3*HH, DD, KPAD);
    convert_pad_k<<<g1((long)3*HH*HH),   256, 0, stream>>>(gru_Whh, Whh16, 3*HH, HH, HH);
    convert_pad_k<<<g1((long)HH*HH),     256, 0, stream>>>(pre_W, pre16, HH, HH, HH);
    convert_pad_k<<<g1((long)HH*KPAD),   256, 0, stream>>>(fp_W, fpW16, HH, DD, KPAD);
    convert_pad_k<<<g1((long)3*HH*HH),   256, 0, stream>>>(inproj_W, inproj16, 3*HH, HH, HH);
    convert_pad_k<<<g1((long)HH*HH),     256, 0, stream>>>(outproj_W, outproj16, HH, HH, HH);
    convert_pad_k<<<g1((long)JJ*SPLHN*SPLK), 256, 0, stream>>>(spl_W1, splW1_16,
                                                               JJ*SPLHN, HH + JDD, SPLK);
    convert_w2_k<<<g1((long)JJ*16*SPLHN), 256, 0, stream>>>(spl_W2, splW2_16);

    // ---- frequency features (rfft.real == cosine transform) ----
    cos_table_k<<<g1((long)FFN*SEEDN), 256, 0, stream>>>(costab);
    freq_k<<<dim3(BB*FFN), 128, 0, stream>>>(poses, costab, freq16);
    build_x16_k<<<g1((long)SEEDN*BB*KPAD), 256, 0, stream>>>(poses, X16);

    // ---- attention branch (chunked over batch), produces mean_q(ctx) ----
    for (int ch = 0; ch < BB / BC; ++ch) {
        launch_gemm(stream, freq16 + (size_t)ch*MC*KPAD, KPAD, fpW16, KPAD,
                    MC, HH, KPAD, fp_b, nullptr, 0, 0, nullptr, 0, feat16c, HH);
        launch_gemm(stream, feat16c, HH, inproj16, HH,
                    MC, 3*HH, HH, inproj_b, nullptr, 0, 0, nullptr, 0, qkv16c, 3*HH);
        attention_k<<<dim3(BC, HEADSN), 256, 0, stream>>>(qkv16c, ch*BC, ctxm);
    }
    convert_pad_k<<<g1((long)BB*HH), 256, 0, stream>>>(ctxm, ctxm16, BB, HH, HH);
    // motion_ctx = mean_q(ctx) @ outproj^T + b   (mean commutes with outproj)
    launch_gemm(stream, ctxm16, HH, outproj16, HH,
                BB, HH, HH, outproj_b, nullptr, 0, 0, motion, HH, nullptr, 0);

    // ---- GRU encoder: 120 sequential steps, gi+gh fused per step ----
    init_state_k<<<g1((long)BB*HH), 256, 0, stream>>>(poses, h, h16, x, x16);
    const dim3 ggru(3 * HH / 128, BB / 64, 2);
    for (int t = 0; t < SEEDN; ++t) {
        gemm_gru2_k<<<ggru, 256, 0, stream>>>(
            X16 + (size_t)t*BB*KPAD, Wih16, gru_bih, gi, KPAD,
            h16, Whh16, gru_bhh, gh, HH);
        gru_gates_k<<<g1((long)BB*HH), 256, 0, stream>>>(gi, gh, h, h16);
    }

    // ---- decoder: 24 sequential steps, SPL fused by dependency depth ----
    JointSet d0 = {{0, 1, 6, 8, 9, 0, 0, 0},  {-1, -1, -1, -1, -1, -1, -1, -1}};
    JointSet d1 = {{2, 3, 7, 10, 11, 0, 0, 0}, {0, 1, 6, 8, 9, -1, -1, -1}};
    JointSet d2 = {{4, 5, 12, 13, 0, 0, 0, 0}, {2, 3, 10, 11, -1, -1, -1, -1}};
    for (int t = 0; t < PREDN; ++t) {
        gemm_gru2_k<<<ggru, 256, 0, stream>>>(
            x16, Wih16, gru_bih, gi, KPAD,
            h16, Whh16, gru_bhh, gh, HH);
        gru_gates_k<<<g1((long)BB*HH), 256, 0, stream>>>(gi, gh, h, h16);
        // hid = relu(h @ pre^T + pre_b) + motion_ctx  -> f16
        launch_gemm(stream, h16, HH, pre16, HH,
                    BB, HH, HH, pre_b, motion, HH, 1, nullptr, 0, hid16, HH);
        spl_depth_k<<<dim3(BB/64, 5), 256, 0, stream>>>(hid16, splW1_16, spl_b1,
                                                        splW2_16, spl_b2, preds16, xdelta, d0);
        spl_depth_k<<<dim3(BB/64, 5), 256, 0, stream>>>(hid16, splW1_16, spl_b1,
                                                        splW2_16, spl_b2, preds16, xdelta, d1);
        spl_depth_k<<<dim3(BB/64, 4), 256, 0, stream>>>(hid16, splW1_16, spl_b1,
                                                        splW2_16, spl_b2, preds16, xdelta, d2);
        x_update_k<<<g1((long)BB*DD), 256, 0, stream>>>(x, x16, xdelta, out, t);
    }
}
